// MultLayer_15522011808176
// MI455X (gfx1250) — matrix operand order repaired and checked by reading, not compiled, not run on hardware
//
#include <hip/hip_runtime.h>

typedef __attribute__((ext_vector_type(16))) __bf16 v16bf;
typedef __attribute__((ext_vector_type(8)))  float  v8f;

union FragB16 { v16bf v; float4 f[2]; };

#define NCH 64     // input channels (K of GEMM)
#define OCW 32     // oc per block
#define MT  64     // spatial positions per block
#define NHW 784    // 28*28
#define IHW 56
#define OCT 128

__global__ __launch_bounds__(256, 1)
void winograd_s2_wmma(const float* __restrict__ x,
                      const float* __restrict__ wgt,
                      const float* __restrict__ bias,
                      float* __restrict__ out)
{
    extern __shared__ __bf16 smem[];
    __bf16* sA = smem;                     // [16][MT][NCH]  bf16, 128KB
    __bf16* sB = smem + 16 * MT * NCH;     // [16][OCW][NCH] bf16, 64KB

    const int tid    = threadIdx.x;
    const int mBase  = blockIdx.x * MT;
    const int ocBase = blockIdx.y * OCW;
    const bool isH2  = (ocBase >= 64);

    // ---------- Phase 1a: fp32 input transform -> bf16 A tiles in LDS ----------
    // T1 = H * P * H (H symmetric);  T2 = H2^T * P * H2.  Both sides are the
    // same 4-point butterfly; only the sign pattern of rows 1,3 differs.
    for (int k = 0; k < 16; ++k) {
        int idx = tid + (k << 8);          // 0..4095 = MT*NCH
        int m_l = idx & 63;
        int c   = idx >> 6;
        int m   = mBase + m_l;
        int n   = m / NHW;
        int hw  = m - n * NHW;
        int h   = hw / 28;
        int w   = hw - h * 28;
        const float* xb = x + ((n * NCH + c) * IHW) * IHW;

        float p[4][4];                     // p[a][b]: a = w-offset, b = h-offset
        #pragma unroll
        for (int b = 0; b < 4; ++b) {
            int row = 2 * h - 1 + b;
            bool rok = (unsigned)row < (unsigned)IHW;
            #pragma unroll
            for (int a = 0; a < 4; ++a) {
                int col = 2 * w - 1 + a;
                p[a][b] = (rok && (unsigned)col < (unsigned)IHW)
                              ? xb[row * IHW + col] : 0.0f;
            }
        }
        // right transform over b -> j
        float t[4][4];
        #pragma unroll
        for (int a = 0; a < 4; ++a) {
            float s0 = p[a][0] + p[a][2], s1 = p[a][1] + p[a][3];
            float d0 = p[a][0] - p[a][2], d1 = p[a][1] - p[a][3];
            t[a][0] = s0 + s1;
            t[a][2] = d0 + d1;
            t[a][1] = isH2 ? (d1 - d0) : (s0 - s1);
            t[a][3] = isH2 ? (s1 - s0) : (d0 - d1);
        }
        // left transform over a -> i, write bf16 rows sA[ij][m][c]
        #pragma unroll
        for (int j = 0; j < 4; ++j) {
            float s0 = t[0][j] + t[2][j], s1 = t[1][j] + t[3][j];
            float d0 = t[0][j] - t[2][j], d1 = t[1][j] - t[3][j];
            float T0 = s0 + s1;
            float T2 = d0 + d1;
            float T1 = isH2 ? (d1 - d0) : (s0 - s1);
            float T3 = isH2 ? (s1 - s0) : (d0 - d1);
            sA[((0 * 4 + j) * MT + m_l) * NCH + c] = (__bf16)T0;
            sA[((1 * 4 + j) * MT + m_l) * NCH + c] = (__bf16)T1;
            sA[((2 * 4 + j) * MT + m_l) * NCH + c] = (__bf16)T2;
            sA[((3 * 4 + j) * MT + m_l) * NCH + c] = (__bf16)T3;
        }
    }

    // ---------- Phase 1b: weight tile -> Bt[ij][oc][c] bf16 in LDS ----------
    for (int k = 0; k < 128; ++k) {
        int e    = tid + (k << 8);         // 0..32767 = 16*OCW*NCH
        int ij   = e >> 11;
        int rem  = e & 2047;
        int oc_l = rem >> 6;
        int c    = rem & 63;
        // weight[o][c][i][j] flat = ((o*64+c)<<4) + (i*4+j)
        float wv = wgt[(((ocBase + oc_l) * NCH + c) << 4) + ij];
        sB[(ij * OCW + oc_l) * NCH + c] = (__bf16)wv;
    }

    __syncthreads();

    // ---------- Phase 2: 16 (i,j) GEMMs via v_wmma_f32_16x16x32_bf16 ----------
    const int lane  = tid & 31;
    const int wave  = tid >> 5;
    const int msub  = wave & 3;            // 4 m-subtiles of 16
    const int osub  = wave >> 2;           // 2 oc-subtiles of 16

    // A fragment (16x32 bf16): lanes 0-15 -> K 0-7 & 16-23 ; lanes 16-31 -> +8
    const int arow  = msub * 16 + (lane & 15);
    const int akoff = (lane >> 4) * 8;
    // B fragment (32x16 bf16): lanes 0-15 -> K 0-7 & 16-23 ; lanes 16-31 -> +8
    const int brow  = osub * 16 + (lane & 15);
    const int bkoff = (lane >> 4) * 8;

    v8f acc[16];
    #pragma unroll
    for (int ij = 0; ij < 16; ++ij)
        acc[ij] = (v8f){0.f, 0.f, 0.f, 0.f, 0.f, 0.f, 0.f, 0.f};

    #pragma unroll
    for (int ij = 0; ij < 16; ++ij) {
        #pragma unroll
        for (int kk = 0; kk < 2; ++kk) {
            const __bf16* ap = sA + (ij * MT + arow) * NCH + kk * 32;
            const __bf16* bp = sB + (ij * OCW + brow) * NCH + kk * 32 + bkoff;
            FragB16 a, b;
            a.f[0] = *(const float4*)(ap + akoff);        // ds_load_b128
            a.f[1] = *(const float4*)(ap + 16 + akoff);
            b.f[0] = *(const float4*)(bp);
            b.f[1] = *(const float4*)(bp + 16);
            acc[ij] = __builtin_amdgcn_wmma_f32_16x16x32_bf16(
                false, a.v, false, b.v, (short)0, acc[ij], false, false);
        }
    }

    // ---------- Phase 3: 2x2 output transform + bias + float2 stores ----------
    const int oc  = ocBase + osub * 16 + (lane & 15);
    const float bv = bias[oc];
    const float sc = 0.0625f;              // 0.25 * 0.25 from IH on both sides

    #pragma unroll
    for (int r = 0; r < 8; ++r) {
        float y[16];
        #pragma unroll
        for (int ij = 0; ij < 16; ++ij) y[ij] = acc[ij][r];

        float u0[4], u1[4];
        #pragma unroll
        for (int j = 0; j < 4; ++j) {
            float a0 = y[0 * 4 + j], a1 = y[1 * 4 + j];
            float a2 = y[2 * 4 + j], a3 = y[3 * 4 + j];
            u0[j] = isH2 ? (a0 + a1 + a2 + a3) : (a0 - a1 + a2 - a3);
            u1[j] = a0 + a1 - a2 - a3;
        }
        float z00 = isH2 ? (u0[0] + u0[1] + u0[2] + u0[3])
                         : (u0[0] - u0[1] + u0[2] - u0[3]);
        float z01 = u0[0] + u0[1] - u0[2] - u0[3];
        float z10 = isH2 ? (u1[0] + u1[1] + u1[2] + u1[3])
                         : (u1[0] - u1[1] + u1[2] - u1[3]);
        float z11 = u1[0] + u1[1] - u1[2] - u1[3];

        int m  = mBase + msub * 16 + ((lane >> 4) * 8 + r);
        int n  = m / NHW;
        int hw = m - n * NHW;
        int h  = hw / 28;
        int w  = hw - h * 28;
        int base = ((n * OCT + oc) * IHW + 2 * h) * IHW + 2 * w;
        *(float2*)(out + base)       = make_float2(z00 * sc + bv, z01 * sc + bv);
        *(float2*)(out + base + IHW) = make_float2(z10 * sc + bv, z11 * sc + bv);
    }
}

extern "C" void kernel_launch(void* const* d_in, const int* in_sizes, int n_in,
                              void* d_out, int out_size, void* d_ws, size_t ws_size,
                              hipStream_t stream) {
    const float* x  = (const float*)d_in[0];   // (32,64,56,56)
    const float* w  = (const float*)d_in[1];   // (128,64,4,4)
    const float* b  = (const float*)d_in[2];   // (1,128,1,1)
    float* out      = (float*)d_out;           // (32,128,56,56)

    // M = 32*28*28 = 25088 -> 392 m-blocks of 64 ; OC=128 -> 4 oc-blocks of 32
    dim3 grid(392, 4, 1);
    dim3 block(256, 1, 1);
    size_t smem = (size_t)(16 * MT * NCH + 16 * OCW * NCH) * 2;  // 192 KB bf16
    winograd_s2_wmma<<<grid, block, smem, stream>>>(x, w, b, out);
}
